// SelfAttentionV1_43130061586975
// MI455X (gfx1250) — matrix operand order repaired and checked by reading, not compiled, not run on hardware
//
#include <hip/hip_runtime.h>
#include <math.h>

// ---------------------------------------------------------------------------
// Self-attention (B=4, S=4096, E=1024, fp32) for MI455X / gfx1250.
// Compute-bound (~0.4 TFLOP vs <1 GB traffic @ 23.3 TB/s): all matrix work on
// v_wmma_f32_16x16x32_bf16 (fp32 accum). Staging uses CDNA5 async copies
// (global_load_async_to_lds_b128, ASYNCcnt) + global_prefetch.
// ---------------------------------------------------------------------------

typedef __attribute__((ext_vector_type(16))) __bf16 v16bf;
typedef __attribute__((ext_vector_type(8)))  float  v8f;
typedef unsigned int   u32;
typedef unsigned short u16;
typedef __attribute__((ext_vector_type(4))) unsigned int u32x4;

#define S_LEN 4096
#define EMB   1024
#define BATCH 4
#define LDSK  40   // LDS row stride in u16 (32 data + 8 pad; 80B = 16B aligned)

__device__ __forceinline__ u16 f2bf(float f) {
  u32 u = __builtin_bit_cast(u32, f);
  u += 0x7fffu + ((u >> 16) & 1u);          // round-to-nearest-even
  return (u16)(u >> 16);
}

// ---- CDNA5 async copy global->LDS (tracked by ASYNCcnt) -------------------
// LDS destination is a 32-bit byte address (low 32 bits of the generic
// shared-space address per the flat aperture rules); global address in a
// VGPR pair ("GV" mode).
__device__ __forceinline__ void async_ld128(u32 lds_addr, const void* gptr) {
  asm volatile("global_load_async_to_lds_b128 %0, %1, off"
               :: "v"(lds_addr), "v"((unsigned long long)(size_t)gptr)
               : "memory");
}
__device__ __forceinline__ void wait_async0() {
  asm volatile("s_wait_asynccnt 0x0" ::: "memory");
}

union FragU { v16bf v; u32x4 q[2]; };

// A fragment (16x32 bf16): lanes 0-15 hold K 0-7 & 16-23, lanes 16-31 K 8-15 & 24-31
__device__ __forceinline__ v16bf load_frag_a(const u16* row, int khalf) {
  FragU f;
  f.q[0] = *(const u32x4*)(row + khalf * 8);
  f.q[1] = *(const u32x4*)(row + 16 + khalf * 8);
  return f.v;
}
// B fragment (32x16 bf16): lane = column; lanes 0-15 K 0-7 & 16-23, lanes 16-31 K 8-15 & 24-31
__device__ __forceinline__ v16bf load_frag_b(const u16* row, int khalf) {
  FragU f;
  f.q[0] = *(const u32x4*)(row + khalf * 8);
  f.q[1] = *(const u32x4*)(row + 16 + khalf * 8);
  return f.v;
}

// ---------------------------------------------------------------------------
// fp32 -> bf16 conversion
// ---------------------------------------------------------------------------
__global__ void cvt_kernel(const float* __restrict__ in, u16* __restrict__ out,
                           size_t n) {
  size_t i = (size_t)blockIdx.x * blockDim.x + threadIdx.x;
  size_t stride = (size_t)gridDim.x * blockDim.x;
  for (; i < n; i += stride) out[i] = f2bf(in[i]);
}

// ---------------------------------------------------------------------------
// Generic bf16 WMMA GEMM:  D = (A @ Bt^T + bias) * scale
//   A  : M x K (row-major, bf16)      Bt : N x K (row-major, bf16)
// Tile 128x128 per workgroup, 8 waves in 2x4 grid, 64x32 per wave
// (4x2 accumulators). Double-buffered LDS fed by async copies.
// Output bf16 (opt transposed) or fp32.
// ---------------------------------------------------------------------------
__global__ __launch_bounds__(256)
void gemm_bf16_kernel(const u16* __restrict__ A, const u16* __restrict__ Bt,
                      const float* __restrict__ bias,
                      u16* __restrict__ outB, float* __restrict__ outF,
                      int K, int lda, int ldb, int ldo,
                      size_t aStride, size_t bStride, size_t oStride,
                      float scale, int outTrans)
{
  __shared__ u16 As[2][128][LDSK];
  __shared__ u16 Bs[2][128][LDSK];

  const int tid = threadIdx.x;
  const int l   = tid & 31;
  const int w   = tid >> 5;
  const int wm  = w >> 2;   // 0..1
  const int wn  = w & 3;    // 0..3

  const size_t aOff = (size_t)blockIdx.z * aStride;
  const size_t bOff = (size_t)blockIdx.z * bStride;
  const size_t oOff = (size_t)blockIdx.z * oStride;

  const int mBase = blockIdx.y * 128;
  const int nBase = blockIdx.x * 128;

  // staging: each thread moves 16 bf16 of A and 16 of B per k-step
  const int srow  = tid >> 1;
  const int skoff = (tid & 1) * 16;
  const u16* aRow = A + aOff + (size_t)(mBase + srow) * lda + skoff;
  const u16* bRow = Bt + bOff + (size_t)(nBase + srow) * ldb + skoff;

  v8f acc[4][2];
#pragma unroll
  for (int mt = 0; mt < 4; ++mt)
#pragma unroll
    for (int nt = 0; nt < 2; ++nt)
#pragma unroll
      for (int r = 0; r < 8; ++r) acc[mt][nt][r] = 0.0f;

  auto stage = [&](int buf, int kt) {
    const u16* ap = aRow + kt * 32;
    const u32 la = (u32)(size_t)&As[buf][srow][skoff];
    async_ld128(la,      ap);
    async_ld128(la + 16, ap + 8);
    const u16* bp = bRow + kt * 32;
    const u32 lb = (u32)(size_t)&Bs[buf][srow][skoff];
    async_ld128(lb,      bp);
    async_ld128(lb + 16, bp + 8);
  };

  const int KT = K >> 5;
  stage(0, 0);
  wait_async0();
  __syncthreads();

  const int khalf = l >> 4;
  const int lr    = l & 15;

  for (int kt = 0; kt < KT; ++kt) {
    const int buf = kt & 1;
    if (kt + 1 < KT) stage(buf ^ 1, kt + 1);
    if (kt + 2 < KT) {
      __builtin_prefetch(aRow + (kt + 2) * 32, 0, 1);
      __builtin_prefetch(bRow + (kt + 2) * 32, 0, 1);
    }

    v16bf aF[4], bF[2];
#pragma unroll
    for (int mt = 0; mt < 4; ++mt)
      aF[mt] = load_frag_a(&As[buf][wm * 64 + mt * 16 + lr][0], khalf);
#pragma unroll
    for (int nt = 0; nt < 2; ++nt)
      bF[nt] = load_frag_b(&Bs[buf][wn * 32 + nt * 16 + lr][0], khalf);

#pragma unroll
    for (int mt = 0; mt < 4; ++mt)
#pragma unroll
      for (int nt = 0; nt < 2; ++nt)
        acc[mt][nt] = __builtin_amdgcn_wmma_f32_16x16x32_bf16(
            false, aF[mt], false, bF[nt], (short)0, acc[mt][nt], false, false);

    wait_async0();
    __syncthreads();
  }

  // Epilogue. C layout: VGPR r, lane l -> row = r + 8*(l>=16), col = l%16
  const int r0 = (l >> 4) * 8;
#pragma unroll
  for (int mt = 0; mt < 4; ++mt) {
#pragma unroll
    for (int nt = 0; nt < 2; ++nt) {
      const int col = nBase + wn * 32 + nt * 16 + lr;
      const float bv = bias ? bias[col] : 0.0f;
#pragma unroll
      for (int r = 0; r < 8; ++r) {
        const int row = mBase + wm * 64 + mt * 16 + r0 + r;
        const float v = (acc[mt][nt][r] + bv) * scale;
        if (outF) {
          outF[oOff + (size_t)row * ldo + col] = v;
        } else if (outTrans) {
          outB[oOff + (size_t)col * ldo + row] = f2bf(v);
        } else {
          outB[oOff + (size_t)row * ldo + col] = f2bf(v);
        }
      }
    }
  }
}

// ---------------------------------------------------------------------------
// Attention score pass. One workgroup = (batch, 64-query tile).
// Iterates key tiles of 64, computing S = Qs @ Ks^T (Q pre-scaled by 1/sqrt(E))
// via WMMA into LDS.
//   mode 0: accumulate running row-max m and row sum-exp l -> stats
//   mode 1: recompute S, write P = exp(S - m)/l as bf16
// 8 waves in 4x2 grid: each wave one 16x32 patch (2 accumulators).
// ---------------------------------------------------------------------------
__global__ __launch_bounds__(256)
void attn_kernel(const u16* __restrict__ Qb, const u16* __restrict__ Kb,
                 float* __restrict__ stats, u16* __restrict__ P, int mode)
{
  __shared__ u16   Qs[2][64][LDSK];
  __shared__ u16   Ks[2][64][LDSK];
  __shared__ float Ssm[64][64];
  __shared__ float mrow[64];
  __shared__ float irow[64];

  const int tid = threadIdx.x;
  const int l   = tid & 31;
  const int w   = tid >> 5;
  const int wm  = w >> 1;   // 0..3
  const int wn  = w & 1;    // 0..1

  const int b     = blockIdx.y;
  const int qBase = blockIdx.x * 64;
  const size_t qRow0 = (size_t)b * S_LEN + qBase;

  const int srow  = tid >> 2;
  const int skoff = (tid & 3) * 8;
  const u16* qPtr = Qb + (qRow0 + srow) * EMB + skoff;

  float m_run = -INFINITY, l_run = 0.0f;
  if (mode == 1 && tid < 64) {
    mrow[tid] = stats[(qRow0 + tid) * 2 + 0];
    irow[tid] = 1.0f / stats[(qRow0 + tid) * 2 + 1];
  }
  __syncthreads();

  const int khalf = l >> 4;
  const int lr    = l & 15;

  for (int jt = 0; jt < S_LEN / 64; ++jt) {
    const u16* kPtr = Kb + ((size_t)b * S_LEN + jt * 64 + srow) * EMB + skoff;

    v8f acc[2];
#pragma unroll
    for (int nt = 0; nt < 2; ++nt)
#pragma unroll
      for (int r = 0; r < 8; ++r) acc[nt][r] = 0.0f;

    auto stage = [&](int buf, int kt) {
      async_ld128((u32)(size_t)&Qs[buf][srow][skoff], qPtr + kt * 32);
      async_ld128((u32)(size_t)&Ks[buf][srow][skoff], kPtr + kt * 32);
    };

    stage(0, 0);
    wait_async0();
    __syncthreads();
    for (int kt = 0; kt < EMB / 32; ++kt) {
      const int buf = kt & 1;
      if (kt + 1 < EMB / 32) stage(buf ^ 1, kt + 1);
      if (kt + 2 < EMB / 32) {
        __builtin_prefetch(qPtr + (kt + 2) * 32, 0, 1);
        __builtin_prefetch(kPtr + (kt + 2) * 32, 0, 1);
      }
      const v16bf aF = load_frag_a(&Qs[buf][wm * 16 + lr][0], khalf);
#pragma unroll
      for (int nt = 0; nt < 2; ++nt) {
        const v16bf bF = load_frag_b(&Ks[buf][wn * 32 + nt * 16 + lr][0], khalf);
        acc[nt] = __builtin_amdgcn_wmma_f32_16x16x32_bf16(
            false, aF, false, bF, (short)0, acc[nt], false, false);
      }
      wait_async0();
      __syncthreads();
    }

    // spill score tile to LDS
    const int r0 = (l >> 4) * 8;
#pragma unroll
    for (int nt = 0; nt < 2; ++nt)
#pragma unroll
      for (int r = 0; r < 8; ++r)
        Ssm[wm * 16 + r0 + r][wn * 32 + nt * 16 + lr] = acc[nt][r];
    __syncthreads();

    if (mode == 0) {
      if (tid < 64) {
        float tmax = -INFINITY;
        for (int j = 0; j < 64; ++j) tmax = fmaxf(tmax, Ssm[tid][j]);
        const float mNew = fmaxf(m_run, tmax);
        float s = 0.0f;
        for (int j = 0; j < 64; ++j) s += __expf(Ssm[tid][j] - mNew);
        l_run = l_run * __expf(m_run - mNew) + s;
        m_run = mNew;
      }
    } else {
      const int row = tid >> 2;
      const int c0  = (tid & 3) * 16;
      const float mv = mrow[row];
      const float iv = irow[row];
      const size_t base = (qRow0 + row) * (size_t)S_LEN + (size_t)jt * 64 + c0;
#pragma unroll
      for (int j = 0; j < 16; ++j)
        P[base + j] = f2bf(__expf(Ssm[row][c0 + j] - mv) * iv);
    }
    __syncthreads();
  }

  if (mode == 0 && tid < 64) {
    stats[(qRow0 + tid) * 2 + 0] = m_run;
    stats[(qRow0 + tid) * 2 + 1] = l_run;
  }
}

// ---------------------------------------------------------------------------
// Host-side orchestration
// ---------------------------------------------------------------------------
extern "C" void kernel_launch(void* const* d_in, const int* in_sizes, int n_in,
                              void* d_out, int out_size, void* d_ws, size_t ws_size,
                              hipStream_t stream)
{
  (void)in_sizes; (void)n_in; (void)out_size; (void)ws_size;

  const float* X  = (const float*)d_in[0];
  const float* Wq = (const float*)d_in[1];
  const float* bq = (const float*)d_in[2];
  const float* Wk = (const float*)d_in[3];
  const float* bk = (const float*)d_in[4];
  const float* Wv = (const float*)d_in[5];
  const float* bv = (const float*)d_in[6];
  float* out = (float*)d_out;

  char* ws = (char*)d_ws;
  size_t off = 0;
  auto alloc = [&](size_t bytes) -> char* {
    char* p = ws + off;
    off += (bytes + 255) & ~(size_t)255;
    return p;
  };

  const size_t nX = (size_t)BATCH * S_LEN * EMB;  // 16.7M
  const size_t nW = (size_t)EMB * EMB;            // 1.05M

  u16* Xb    = (u16*)alloc(nX * 2);
  u16* Wqb   = (u16*)alloc(nW * 2);
  u16* Wkb   = (u16*)alloc(nW * 2);
  u16* Wvb   = (u16*)alloc(nW * 2);
  u16* Qb    = (u16*)alloc(nX * 2);               // pre-scaled by 1/sqrt(E)
  u16* Kb    = (u16*)alloc(nX * 2);
  u16* Vtb   = (u16*)alloc(nX * 2);               // [B][E][S] (transposed)
  float* stats = (float*)alloc((size_t)BATCH * S_LEN * 2 * sizeof(float));
  u16* P     = (u16*)alloc((size_t)BATCH * S_LEN * S_LEN * 2);

  // 1) fp32 -> bf16
  cvt_kernel<<<1024, 256, 0, stream>>>(X,  Xb,  nX);
  cvt_kernel<<<256,  256, 0, stream>>>(Wq, Wqb, nW);
  cvt_kernel<<<256,  256, 0, stream>>>(Wk, Wkb, nW);
  cvt_kernel<<<256,  256, 0, stream>>>(Wv, Wvb, nW);

  const float scale = 1.0f / 32.0f;  // 1/sqrt(1024), folded into Q

  // 2) Q = (X @ Wq^T + bq) * scale   [16384 x 1024]
  gemm_bf16_kernel<<<dim3(EMB / 128, (BATCH * S_LEN) / 128, 1), 256, 0, stream>>>(
      Xb, Wqb, bq, Qb, nullptr, EMB, EMB, EMB, EMB, 0, 0, 0, scale, 0);
  // 3) K = X @ Wk^T + bk
  gemm_bf16_kernel<<<dim3(EMB / 128, (BATCH * S_LEN) / 128, 1), 256, 0, stream>>>(
      Xb, Wkb, bk, Kb, nullptr, EMB, EMB, EMB, EMB, 0, 0, 0, 1.0f, 0);
  // 4) V = X @ Wv^T + bv, stored transposed per batch: Vt[b][e][s]
  gemm_bf16_kernel<<<dim3(EMB / 128, S_LEN / 128, BATCH), 256, 0, stream>>>(
      Xb, Wvb, bv, Vtb, nullptr, EMB, EMB, EMB, S_LEN,
      (size_t)S_LEN * EMB, 0, (size_t)EMB * S_LEN, 1.0f, 1);

  // 5) softmax stats (row max + sum-exp) over S = Q @ K^T
  attn_kernel<<<dim3(S_LEN / 64, BATCH), 256, 0, stream>>>(Qb, Kb, stats, P, 0);
  // 6) P = softmax(S) as bf16
  attn_kernel<<<dim3(S_LEN / 64, BATCH), 256, 0, stream>>>(Qb, Kb, stats, P, 1);

  // 7) out = P @ V   (A = P [S x S], Bt = Vt [E x S]) -> fp32
  gemm_bf16_kernel<<<dim3(EMB / 128, S_LEN / 128, BATCH), 256, 0, stream>>>(
      P, Vtb, nullptr, nullptr, out, S_LEN, S_LEN, S_LEN, EMB,
      (size_t)S_LEN * S_LEN, (size_t)EMB * S_LEN, (size_t)S_LEN * EMB, 1.0f, 0);
}
